// WordInputCombiner_2929167696464
// MI455X (gfx1250) — matrix operand order repaired and checked by reading, not compiled, not run on hardware
//
#include <hip/hip_runtime.h>
#include <hip/hip_bf16.h>
#include <stdint.h>

// ---------------------------------------------------------------------------
// Problem constants (from reference setup_inputs; word-span pattern is fixed:
// [F,F,F,E,O,F,E,O] -> 2 words per 8 tokens -> NWORD=1024, SWORD(maxL)=6)
// ---------------------------------------------------------------------------
#define BSZ    16
#define SLEN   256
#define DMODEL 1024
#define NHEAD  16
#define HDIM   64
#define NLAYER 2
#define FFDIM  4096
#define NWORD  1024
#define SWORD  6
#define MCTX   (BSZ*SLEN)      // 4096 rows
#define MWORD  (NWORD*SWORD)   // 6144 rows

typedef __attribute__((ext_vector_type(16))) __bf16 v16bf;
typedef __attribute__((ext_vector_type(8)))  float  v8f;

union FragBF { v16bf v; unsigned int p[8]; };

__device__ __forceinline__ unsigned short f2bf(float f) {
  unsigned int u = __float_as_uint(f);
  unsigned int r = u + 0x7FFFu + ((u >> 16) & 1u);   // round-to-nearest-even
  return (unsigned short)(r >> 16);
}
__device__ __forceinline__ unsigned int pack2bf(float lo, float hi) {
  return (unsigned int)f2bf(lo) | ((unsigned int)f2bf(hi) << 16);
}

// ---------------------------------------------------------------------------
// bf16-WMMA GEMM: C[M,N] = A[M,K] @ W[K,N] + bias (+optional relu), fp32 I/O.
// Block: 256 threads = 8 waves; block tile 128x128; wave tile 32x64
// (2x4 = 8 x v_wmma_f32_16x16x32_bf16 per wave per K-step of 32).
// LDS holds bf16 pairs packed in uint32, laid out to match the ISA 7.12.2
// 16-bit fragment layouts so fragment gathers are aligned b64 loads.
// ---------------------------------------------------------------------------
#define BM   128
#define BN   128
#define BKK  32
#define LPAD 2      // row stride 16+2=18 uints: 8B-aligned, bank-spread

__global__ __launch_bounds__(256) void gemm_bf16_wmma_kernel(
    const float* __restrict__ A, const float* __restrict__ W,
    const float* __restrict__ bias, float* __restrict__ C,
    int M, int N, int K, int relu)
{
  // packed pairs along K: Asp[m][k/2], Bsp[n][k/2] ({lo=K even, hi=K odd})
  __shared__ unsigned int Asp[BM][BKK/2 + LPAD];
  __shared__ unsigned int Bsp[BN][BKK/2 + LPAD];

  const int tid   = threadIdx.x;
  const int lane  = tid & 31;
  const int wid   = tid >> 5;        // 0..7
  const int waveM = wid >> 1;        // 0..3 -> 32-row strip
  const int waveN = wid & 1;         // 0..1 -> 64-col strip
  const int half  = lane >> 4;       // ISA: lanes 16-31 take k 8-15 and 24-31
  const int l16   = lane & 15;

  v8f acc[2][4];
  #pragma unroll
  for (int a = 0; a < 2; ++a)
    #pragma unroll
    for (int b = 0; b < 4; ++b)
      #pragma unroll
      for (int i = 0; i < 8; ++i) acc[a][b][i] = 0.0f;

  const int rowBase = blockIdx.y * BM;
  const int colBase = blockIdx.x * BN;

  for (int k0 = 0; k0 < K; k0 += BKK) {
    if (k0 + BKK < K) {   // prefetch next K-step tiles (global_prefetch_b8)
      __builtin_prefetch(A + (size_t)(rowBase + (tid >> 1)) * K + k0 + BKK + (tid & 1) * 16, 0, 1);
      __builtin_prefetch(W + (size_t)(k0 + BKK + (tid >> 3)) * N + colBase + (tid & 7) * 16, 0, 1);
    }

    // ---- stage A tile: 128x32 fp32, 1024 float4 over 256 threads,
    //      packed as one b64 LDS store per float4 ----
    #pragma unroll
    for (int i = 0; i < 4; ++i) {
      int s  = tid + i * 256;
      int m  = s >> 3;              // 0..127
      int kq = (s & 7) << 2;        // 0,4,..,28
      const float4 f = *(const float4*)(A + (size_t)(rowBase + m) * K + k0 + kq);
      uint2 pk;
      pk.x = pack2bf(f.x, f.y);
      pk.y = pack2bf(f.z, f.w);
      *(uint2*)&Asp[m][kq >> 1] = pk;
    }
    // ---- stage W tile: 32x128 fp32 as 2x2 (k x n) micro-blocks:
    //      two coalesced float2 row reads -> two packed b32 LDS stores ----
    #pragma unroll
    for (int i = 0; i < 4; ++i) {
      int s  = tid + i * 256;       // 0..1023 units
      int nu = s & 63;              // n-pair 0..63
      int ku = s >> 6;              // k-pair 0..15
      const float2 r0 = *(const float2*)(W + (size_t)(k0 + 2 * ku + 0) * N + colBase + 2 * nu);
      const float2 r1 = *(const float2*)(W + (size_t)(k0 + 2 * ku + 1) * N + colBase + 2 * nu);
      Bsp[2 * nu + 0][ku] = pack2bf(r0.x, r1.x);
      Bsp[2 * nu + 1][ku] = pack2bf(r0.y, r1.y);
    }
    __syncthreads();

    // ---- fragment gathers (aligned b64 LDS loads) ----
    // A VGPR v <-> pair (v/4)*8 + half*4 + v%4: pairs are contiguous runs of 4
    FragBF af[2];
    #pragma unroll
    for (int tm = 0; tm < 2; ++tm) {
      int m = waveM * 32 + tm * 16 + l16;
      uint2 q0 = *(const uint2*)&Asp[m][half * 4 + 0];
      uint2 q1 = *(const uint2*)&Asp[m][half * 4 + 2];
      uint2 q2 = *(const uint2*)&Asp[m][8 + half * 4 + 0];
      uint2 q3 = *(const uint2*)&Asp[m][8 + half * 4 + 2];
      af[tm].p[0] = q0.x; af[tm].p[1] = q0.y; af[tm].p[2] = q1.x; af[tm].p[3] = q1.y;
      af[tm].p[4] = q2.x; af[tm].p[5] = q2.y; af[tm].p[6] = q3.x; af[tm].p[7] = q3.y;
    }
    // B VGPR v <-> pair (v/4)*8 + half*4 + v%4: same order as A
    FragBF bfr[4];
    #pragma unroll
    for (int tn = 0; tn < 4; ++tn) {
      int n = waveN * 64 + tn * 16 + l16;
      uint2 r0 = *(const uint2*)&Bsp[n][half * 4 + 0];
      uint2 r1 = *(const uint2*)&Bsp[n][half * 4 + 2];
      uint2 r2 = *(const uint2*)&Bsp[n][8 + half * 4 + 0];
      uint2 r3 = *(const uint2*)&Bsp[n][8 + half * 4 + 2];
      bfr[tn].p[0] = r0.x; bfr[tn].p[1] = r0.y; bfr[tn].p[2] = r1.x; bfr[tn].p[3] = r1.y;
      bfr[tn].p[4] = r2.x; bfr[tn].p[5] = r2.y; bfr[tn].p[6] = r3.x; bfr[tn].p[7] = r3.y;
    }

    #pragma unroll
    for (int tm = 0; tm < 2; ++tm)
      #pragma unroll
      for (int tn = 0; tn < 4; ++tn)
        acc[tm][tn] = __builtin_amdgcn_wmma_f32_16x16x32_bf16(
            false, af[tm].v, false, bfr[tn].v, (short)0, acc[tm][tn], false, false);
    __syncthreads();
  }

  // ---- epilogue: bias (+relu), fp32 store. D layout: M = r + 8*half, N = l16 ----
  #pragma unroll
  for (int tm = 0; tm < 2; ++tm) {
    #pragma unroll
    for (int tn = 0; tn < 4; ++tn) {
      int n = colBase + waveN * 64 + tn * 16 + l16;
      float bv = bias ? bias[n] : 0.0f;
      #pragma unroll
      for (int r = 0; r < 8; ++r) {
        int m = rowBase + waveM * 32 + tm * 16 + (half << 3) + r;
        float val = acc[tm][tn][r] + bv;
        if (relu) val = fmaxf(val, 0.0f);
        C[(size_t)m * N + n] = val;
      }
    }
  }
}

// ---------------------------------------------------------------------------
// Context attention: one block per (b,h,q); 256 threads = one score each,
// LDS softmax, then 64 threads do A@V.
// ---------------------------------------------------------------------------
__global__ __launch_bounds__(256) void attn_ctx_kernel(
    const float* __restrict__ Q, const float* __restrict__ K,
    const float* __restrict__ V, float* __restrict__ O,
    const int* __restrict__ lengths)
{
  __shared__ float sc[SLEN];
  __shared__ float red[SLEN];
  const int idx = blockIdx.x;
  const int q   = idx % SLEN;
  const int bh  = idx / SLEN;
  const int h   = bh % NHEAD;
  const int b   = bh / NHEAD;
  const int tid = threadIdx.x;
  const int len = lengths[b];
  const float scale = 0.125f;  // 1/sqrt(64)

  const float* qv = Q + ((size_t)(b * SLEN + q))   * DMODEL + h * HDIM;
  const float* kv = K + ((size_t)(b * SLEN + tid)) * DMODEL + h * HDIM;
  float s = 0.0f;
  #pragma unroll 8
  for (int d = 0; d < HDIM; ++d) s += qv[d] * kv[d];
  s *= scale;
  if (tid >= len) s = -1e9f;   // src_key_padding_mask

  red[tid] = s; __syncthreads();
  for (int st = 128; st > 0; st >>= 1) { if (tid < st) red[tid] = fmaxf(red[tid], red[tid + st]); __syncthreads(); }
  float mx = red[0]; __syncthreads();
  float e = __expf(s - mx);
  red[tid] = e; __syncthreads();
  for (int st = 128; st > 0; st >>= 1) { if (tid < st) red[tid] += red[tid + st]; __syncthreads(); }
  float denom = red[0]; __syncthreads();
  sc[tid] = e / denom;
  __syncthreads();

  if (tid < HDIM) {
    float o = 0.0f;
    const float* vp = V + ((size_t)(b * SLEN)) * DMODEL + h * HDIM + tid;
    for (int kk = 0; kk < SLEN; ++kk) o += sc[kk] * vp[(size_t)kk * DMODEL];
    O[((size_t)(b * SLEN + q)) * DMODEL + h * HDIM + tid] = o;
  }
}

// ---------------------------------------------------------------------------
// Word attention: S'=6 is tiny -> one thread per (w,h,q).
// ---------------------------------------------------------------------------
__global__ void attn_word_kernel(
    const float* __restrict__ Q, const float* __restrict__ K,
    const float* __restrict__ V, float* __restrict__ O,
    const int* __restrict__ wlen)
{
  int t = blockIdx.x * blockDim.x + threadIdx.x;
  if (t >= NWORD * NHEAD * SWORD) return;
  int q = t % SWORD; int r = t / SWORD;
  int h = r % NHEAD; int w = r / NHEAD;
  int len = wlen[w];
  const float scale = 0.125f;
  const float* qv = Q + ((size_t)(w * SWORD + q)) * DMODEL + h * HDIM;

  float sc[SWORD];
  float mx = -1e30f;
  #pragma unroll
  for (int kk = 0; kk < SWORD; ++kk) {
    const float* kv = K + ((size_t)(w * SWORD + kk)) * DMODEL + h * HDIM;
    float s = 0.0f;
    for (int d = 0; d < HDIM; ++d) s += qv[d] * kv[d];
    s *= scale;
    if (kk >= len) s = -1e9f;
    sc[kk] = s; mx = fmaxf(mx, s);
  }
  float denom = 0.0f;
  #pragma unroll
  for (int kk = 0; kk < SWORD; ++kk) { sc[kk] = __expf(sc[kk] - mx); denom += sc[kk]; }
  float inv = 1.0f / denom;
  for (int d = 0; d < HDIM; ++d) {
    float o = 0.0f;
    #pragma unroll
    for (int kk = 0; kk < SWORD; ++kk)
      o += sc[kk] * V[((size_t)(w * SWORD + kk)) * DMODEL + h * HDIM + d];
    O[((size_t)(w * SWORD + q)) * DMODEL + h * HDIM + d] = o * inv;
  }
}

// ---------------------------------------------------------------------------
// out = LayerNorm(x + res)*g + b ; one block per row, D=1024, in-place safe.
// ---------------------------------------------------------------------------
__global__ __launch_bounds__(256) void ln_residual_kernel(
    const float* __restrict__ x, const float* __restrict__ res,
    const float* __restrict__ g, const float* __restrict__ b,
    float* __restrict__ out)
{
  __shared__ float red[256];
  const int row = blockIdx.x;
  const int tid = threadIdx.x;
  const float* xr = x   + (size_t)row * DMODEL;
  const float* rr = res + (size_t)row * DMODEL;
  float v[4]; float s = 0.0f;
  #pragma unroll
  for (int i = 0; i < 4; ++i) { v[i] = xr[tid + i * 256] + rr[tid + i * 256]; s += v[i]; }
  red[tid] = s; __syncthreads();
  for (int st = 128; st > 0; st >>= 1) { if (tid < st) red[tid] += red[tid + st]; __syncthreads(); }
  float mean = red[0] * (1.0f / DMODEL); __syncthreads();
  float s2 = 0.0f;
  #pragma unroll
  for (int i = 0; i < 4; ++i) { float d = v[i] - mean; s2 += d * d; }
  red[tid] = s2; __syncthreads();
  for (int st = 128; st > 0; st >>= 1) { if (tid < st) red[tid] += red[tid + st]; __syncthreads(); }
  float inv = rsqrtf(red[0] * (1.0f / DMODEL) + 1e-5f); __syncthreads();
  #pragma unroll
  for (int i = 0; i < 4; ++i) {
    int c = tid + i * 256;
    out[(size_t)row * DMODEL + c] = (v[i] - mean) * inv * g[c] + b[c];
  }
}

// ---------------------------------------------------------------------------
// X[b,s,:] = encoded[b,s,:] + pos_emb[s,:]
// ---------------------------------------------------------------------------
__global__ __launch_bounds__(256) void add_pos_kernel(
    const float* __restrict__ enc, const float* __restrict__ pos,
    float* __restrict__ X)
{
  int row = blockIdx.x;             // b*SLEN + s
  int s = row % SLEN;
  #pragma unroll
  for (int i = 0; i < 4; ++i) {
    int d = threadIdx.x + i * 256;
    X[(size_t)row * DMODEL + d] = enc[(size_t)row * DMODEL + d] + pos[(size_t)s * DMODEL + d];
  }
}

// ---------------------------------------------------------------------------
// Word-span map from combine_labels (FRONT=1, END=2): per-row sequential scan.
// ---------------------------------------------------------------------------
__global__ void build_word_map_kernel(
    const int* __restrict__ labels,
    int* __restrict__ tok_word, int* __restrict__ tok_slot, int* __restrict__ wlenA)
{
  __shared__ int ends[32];
  __shared__ int base[32];
  int b = threadIdx.x;
  if (b < BSZ) {
    int c = 0;
    for (int s = 0; s < SLEN; ++s) c += (labels[b * SLEN + s] == 2);
    ends[b] = c;
  }
  __syncthreads();
  if (threadIdx.x == 0) {
    int acc = 0;
    for (int i = 0; i < BSZ; ++i) { base[i] = acc; acc += ends[i]; }
  }
  __syncthreads();
  if (b < BSZ) {
    int w = base[b]; int frontrun = 0;
    for (int s = 0; s < SLEN; ++s) {
      int t = b * SLEN + s;
      int lab = labels[t];
      if (lab == 1) {                       // FRONT
        frontrun++; tok_word[t] = w; tok_slot[t] = frontrun;
      } else if (lab == 2) {                // END
        tok_word[t] = w; tok_slot[t] = frontrun + 1;
        wlenA[w] = frontrun + 3;            // subwords + BOS + EOS
        w++; frontrun = 0;
      } else {
        tok_word[t] = -1; tok_slot[t] = 0; frontrun = 0;
      }
    }
  }
}

// Xw[w,s,:] = pos[s] + (s==0 ? BOS : s==wlen-1 ? EOS : 0); tokens overwritten later
__global__ __launch_bounds__(256) void fill_word_base_kernel(
    float* __restrict__ Xw, const float* __restrict__ pos,
    const float* __restrict__ special, const int* __restrict__ wlenA)
{
  int ws = blockIdx.x;
  int s = ws % SWORD; int w = ws / SWORD;
  int len = wlenA[w];
  #pragma unroll
  for (int i = 0; i < 4; ++i) {
    int d = threadIdx.x + i * 256;
    float v = pos[(size_t)s * DMODEL + d];
    if (s == 0)            v += special[0 * DMODEL + d];  // BOS_ID=0
    else if (s == len - 1) v += special[1 * DMODEL + d];  // EOS_ID=1
    Xw[((size_t)w * SWORD + s) * DMODEL + d] = v;
  }
}

// Xw[word,slot,:] = X[token,:] + pos[slot,:]
__global__ __launch_bounds__(256) void scatter_tokens_kernel(
    const float* __restrict__ X, const float* __restrict__ pos,
    const int* __restrict__ tok_word, const int* __restrict__ tok_slot,
    float* __restrict__ Xw)
{
  int t = blockIdx.x;
  int w = tok_word[t];
  if (w < 0) return;
  int s = tok_slot[t];
  #pragma unroll
  for (int i = 0; i < 4; ++i) {
    int d = threadIdx.x + i * 256;
    Xw[((size_t)w * SWORD + s) * DMODEL + d] =
        X[(size_t)t * DMODEL + d] + pos[(size_t)s * DMODEL + d];
  }
}

// out[w,:] = Xw[w,0,:]  (BOS representation)
__global__ __launch_bounds__(256) void copy_bos_kernel(
    const float* __restrict__ Xw, float* __restrict__ out)
{
  int w = blockIdx.x;
  #pragma unroll
  for (int i = 0; i < 4; ++i) {
    int d = threadIdx.x + i * 256;
    out[(size_t)w * DMODEL + d] = Xw[((size_t)w * SWORD) * DMODEL + d];
  }
}

// ---------------------------------------------------------------------------
// Host-side orchestration
// ---------------------------------------------------------------------------
struct EncW {
  const float *Wq, *Wk, *Wv, *Wo, *bq, *bk, *bv, *bo, *b2, *be1, *be2;
  const float *W1, *b1, *W2, *g1, *g2;
};

static void run_layer(float* X, int M, const EncW& w, int layer,
                      float* Q, float* Kb, float* V, float* AO, float* T, float* H,
                      const int* ctx_lens, const int* wlen, bool isWord,
                      hipStream_t stream)
{
  const size_t dd = (size_t)DMODEL * DMODEL;
  const float* Wq = w.Wq + (size_t)layer * dd;
  const float* Wk = w.Wk + (size_t)layer * dd;
  const float* Wv = w.Wv + (size_t)layer * dd;
  const float* Wo = w.Wo + (size_t)layer * dd;
  const float* W1 = w.W1 + (size_t)layer * DMODEL * FFDIM;
  const float* W2 = w.W2 + (size_t)layer * FFDIM * DMODEL;
  const float* bq = w.bq + (size_t)layer * DMODEL;
  const float* bk = w.bk + (size_t)layer * DMODEL;
  const float* bv = w.bv + (size_t)layer * DMODEL;
  const float* bo = w.bo + (size_t)layer * DMODEL;
  const float* b1 = w.b1 + (size_t)layer * FFDIM;
  const float* b2 = w.b2 + (size_t)layer * DMODEL;
  const float* g1 = w.g1 + (size_t)layer * DMODEL;
  const float* g2 = w.g2 + (size_t)layer * DMODEL;
  const float* be1 = w.be1 + (size_t)layer * DMODEL;
  const float* be2 = w.be2 + (size_t)layer * DMODEL;

  dim3 blk(256);
  dim3 gD(DMODEL / BN, M / BM);
  dim3 gF(FFDIM / BN, M / BM);

  gemm_bf16_wmma_kernel<<<gD, blk, 0, stream>>>(X, Wq, bq, Q,  M, DMODEL, DMODEL, 0);
  gemm_bf16_wmma_kernel<<<gD, blk, 0, stream>>>(X, Wk, bk, Kb, M, DMODEL, DMODEL, 0);
  gemm_bf16_wmma_kernel<<<gD, blk, 0, stream>>>(X, Wv, bv, V,  M, DMODEL, DMODEL, 0);

  if (!isWord) {
    attn_ctx_kernel<<<BSZ * NHEAD * SLEN, 256, 0, stream>>>(Q, Kb, V, AO, ctx_lens);
  } else {
    int tot = NWORD * NHEAD * SWORD;
    attn_word_kernel<<<(tot + 255) / 256, 256, 0, stream>>>(Q, Kb, V, AO, wlen);
  }

  gemm_bf16_wmma_kernel<<<gD, blk, 0, stream>>>(AO, Wo, bo, T, M, DMODEL, DMODEL, 0);
  ln_residual_kernel<<<M, 256, 0, stream>>>(X, T, g1, be1, X);

  gemm_bf16_wmma_kernel<<<gF, blk, 0, stream>>>(X, W1, b1, H, M, FFDIM, DMODEL, 1);
  gemm_bf16_wmma_kernel<<<gD, blk, 0, stream>>>(H, W2, b2, T, M, DMODEL, FFDIM, 0);
  ln_residual_kernel<<<M, 256, 0, stream>>>(X, T, g2, be2, X);
}

extern "C" void kernel_launch(void* const* d_in, const int* in_sizes, int n_in,
                              void* d_out, int out_size, void* d_ws, size_t ws_size,
                              hipStream_t stream)
{
  (void)in_sizes; (void)n_in; (void)out_size; (void)ws_size;

  const float* encoded = (const float*)d_in[0];
  const int*   lengths = (const int*)d_in[1];
  const int*   labels  = (const int*)d_in[2];
  // d_in[3] = lang_id (unused by the forward math)
  const float* pos     = (const float*)d_in[4];
  const float* special = (const float*)d_in[5];

  // setup_inputs dict order: Wq,Wk,Wv,Wo, bq,bk,bv,bo,b2,be1,be2, W1,b1,W2,g1,g2
  EncW ctx = {
    (const float*)d_in[6],  (const float*)d_in[7],  (const float*)d_in[8],  (const float*)d_in[9],
    (const float*)d_in[10], (const float*)d_in[11], (const float*)d_in[12], (const float*)d_in[13],
    (const float*)d_in[14], (const float*)d_in[15], (const float*)d_in[16],
    (const float*)d_in[17], (const float*)d_in[18], (const float*)d_in[19],
    (const float*)d_in[20], (const float*)d_in[21]
  };
  EncW wc = {
    (const float*)d_in[22], (const float*)d_in[23], (const float*)d_in[24], (const float*)d_in[25],
    (const float*)d_in[26], (const float*)d_in[27], (const float*)d_in[28], (const float*)d_in[29],
    (const float*)d_in[30], (const float*)d_in[31], (const float*)d_in[32],
    (const float*)d_in[33], (const float*)d_in[34], (const float*)d_in[35],
    (const float*)d_in[36], (const float*)d_in[37]
  };

  // Workspace layout (floats): 11 regions of S1 = MWORD*DMODEL, then int maps.
  float* ws = (float*)d_ws;
  const size_t S1 = (size_t)MWORD * DMODEL;
  float* X  = ws;            // activations (ctx uses first MCTX rows)
  float* Q  = ws + 1 * S1;
  float* Kb = ws + 2 * S1;
  float* V  = ws + 3 * S1;
  float* AO = ws + 4 * S1;
  float* T  = ws + 5 * S1;
  float* H  = ws + 6 * S1;   // FFN hidden: MWORD*FFDIM = 4*S1
  float* XW = ws + 10 * S1;  // word-encoder activations
  int* ip       = (int*)(ws + 11 * S1);
  int* tok_word = ip;                 // MCTX
  int* tok_slot = ip + MCTX;          // MCTX
  int* wlenA    = ip + 2 * MCTX;      // NWORD

  // -------- context encoder --------
  add_pos_kernel<<<MCTX, 256, 0, stream>>>(encoded, pos, X);
  for (int i = 0; i < NLAYER; ++i)
    run_layer(X, MCTX, ctx, i, Q, Kb, V, AO, T, H, lengths, nullptr, false, stream);

  // -------- word-span gather --------
  build_word_map_kernel<<<1, 32, 0, stream>>>(labels, tok_word, tok_slot, wlenA);
  fill_word_base_kernel<<<NWORD * SWORD, 256, 0, stream>>>(XW, pos, special, wlenA);
  scatter_tokens_kernel<<<MCTX, 256, 0, stream>>>(X, pos, tok_word, tok_slot, XW);

  // -------- word encoder --------
  for (int i = 0; i < NLAYER; ++i)
    run_layer(XW, MWORD, wc, i, Q, Kb, V, AO, T, H, nullptr, wlenA, true, stream);

  // -------- output: BOS position --------
  copy_bos_kernel<<<NWORD, 256, 0, stream>>>(XW, (float*)d_out);
}
